// BLTSelfAttention_15272903704759
// MI455X (gfx1250) — matrix operand order repaired and checked by reading, not compiled, not run on hardware
//
#include <hip/hip_runtime.h>

// ---------------------------------------------------------------- constants
constexpr int B  = 2;
constexpr int S  = 2048;
constexpr int HID = 2048;
constexpr int NH = 16;
constexpr int HD = 128;            // head dim
constexpr int HALF = HD / 2;       // 64 rope freqs
constexpr int M_ROWS = B * S;      // 4096 GEMM rows

constexpr int KPAD = 136;          // padded K-tile row (elems): 272B = 17*16B

typedef __attribute__((ext_vector_type(16))) __bf16 v16bf;
typedef __attribute__((ext_vector_type(8)))  __bf16 v8bf;
typedef __attribute__((ext_vector_type(8)))  float  v8f;

__device__ __forceinline__ v8f wmma_bf16(v16bf a, v16bf b, v8f c) {
  return __builtin_amdgcn_wmma_f32_16x16x32_bf16(false, a, false, b,
                                                 (short)0, c, false, false);
}

__device__ __forceinline__ v16bf cat8(v8bf lo, v8bf hi) {
  return __builtin_shufflevector(lo, hi, 0,1,2,3,4,5,6,7,8,9,10,11,12,13,14,15);
}

// ---------------------------------------------------------------- fp32 -> bf16
__global__ __launch_bounds__(256) void cvt_f32_bf16(const float* __restrict__ in,
                                                    __bf16* __restrict__ out,
                                                    size_t n) {
  size_t i = (size_t)blockIdx.x * blockDim.x + threadIdx.x;
  size_t stride = (size_t)gridDim.x * blockDim.x;
  for (; i < n; i += stride) out[i] = (__bf16)in[i];
}

// ---------------------------------------------------------------- GEMM Y = X * W^T
// X: (4096 x 2048) bf16 row-major, W: (2048 x 2048) bf16 row-major.
// One wave -> 16x128 output tile (A fragment reused 8x).
template <typename OUT>
__global__ __launch_bounds__(256) void gemm_xwt(const __bf16* __restrict__ X,
                                                const __bf16* __restrict__ W,
                                                OUT* __restrict__ Y) {
  constexpr int N = HID, K = HID;
  const int wave = (blockIdx.x * 256 + threadIdx.x) >> 5;
  const int lane = threadIdx.x & 31;
  const int l16  = lane & 15;
  const int hi   = lane >> 4;                 // which half-wave
  const int wpn  = N / 128;                   // 16 waves per row band
  const int m0   = (wave / wpn) * 16;
  const int n0   = (wave % wpn) * 128;

  const __bf16* xrow = X + (size_t)(m0 + l16) * K;

  v8f zero = {0.f,0.f,0.f,0.f,0.f,0.f,0.f,0.f};
  v8f acc[8] = {zero, zero, zero, zero, zero, zero, zero, zero};

  for (int k0 = 0; k0 < K; k0 += 32) {
    __builtin_prefetch(xrow + k0 + 128, 0, 0);   // global_prefetch_b8, A stream
    // A fragment: row m0+l16, K = {8*hi..8*hi+7} and {16+8*hi..16+8*hi+7}
    v8bf a0 = *(const v8bf*)(xrow + k0 + 8 * hi);
    v8bf a1 = *(const v8bf*)(xrow + k0 + 16 + 8 * hi);
    v16bf a = cat8(a0, a1);
#pragma unroll
    for (int t = 0; t < 8; t++) {
      // B fragment: col n0+16t+l16 of W^T == row of W, K = {8*hi..8*hi+7} and {16+8*hi..16+8*hi+7}
      const __bf16* wrow = W + (size_t)(n0 + 16 * t + l16) * K + k0 + 8 * hi;
      v16bf bf = cat8(*(const v8bf*)wrow, *(const v8bf*)(wrow + 16));
      acc[t] = wmma_bf16(a, bf, acc[t]);
    }
  }
#pragma unroll
  for (int t = 0; t < 8; t++)
#pragma unroll
    for (int r = 0; r < 8; r++) {
      size_t row = (size_t)(m0 + r + 8 * hi);
      size_t col = (size_t)(n0 + 16 * t + l16);
      Y[row * N + col] = (OUT)acc[t][r];
    }
}

// ---------------------------------------------------------------- RoPE (in-place, Q and K)
__global__ __launch_bounds__(256) void rope_qk(__bf16* __restrict__ Qb,
                                               __bf16* __restrict__ Kb,
                                               const float* __restrict__ cosp,
                                               const float* __restrict__ sinp) {
  size_t t = (size_t)blockIdx.x * 256 + threadIdx.x;   // B*S*NH*HALF threads
  int    i  = (int)(t & (HALF - 1));
  int    h  = (int)((t >> 6) & (NH - 1));
  size_t bs = t >> 10;                                  // b*S + s  (0..4095)
  float c  = cosp[bs * HALF + i];
  float si = sinp[bs * HALF + i];
  size_t base = bs * HID + (size_t)h * HD + 2 * i;
  float qr = (float)Qb[base], qi = (float)Qb[base + 1];
  Qb[base]     = (__bf16)(c * qr - si * qi);
  Qb[base + 1] = (__bf16)(si * qr + c * qi);
  float kr = (float)Kb[base], ki = (float)Kb[base + 1];
  Kb[base]     = (__bf16)(c * kr - si * ki);
  Kb[base + 1] = (__bf16)(si * kr + c * ki);
}

// ---------------------------------------------------------------- V -> V^T  [b][h][d][s]
__global__ __launch_bounds__(256) void transpose_v(const __bf16* __restrict__ V,
                                                   __bf16* __restrict__ Vt) {
  __shared__ __bf16 tile[32][33];
  const int bh = blockIdx.z;                 // b*NH + h
  const int s0 = blockIdx.x * 32;
  const int d0 = blockIdx.y * 32;
  const int tx = threadIdx.x & 31;
  const int ty = threadIdx.x >> 5;           // 0..7
#pragma unroll
  for (int i = 0; i < 32; i += 8) {
    int s = s0 + ty + i, d = d0 + tx;
    int b = bh >> 4, h = bh & 15;
    tile[ty + i][tx] = V[(size_t)(b * S + s) * HID + h * HD + d];
  }
  __syncthreads();
#pragma unroll
  for (int i = 0; i < 32; i += 8) {
    int d = d0 + ty + i, s = s0 + tx;
    Vt[((size_t)bh * HD + d) * S + s] = tile[tx][ty + i];
  }
}

// ---------------------------------------------------------------- flash attention
// One wave per (b, h, 16-row q tile).  kv processed in blocks of 32.
// K tiles double-buffered in LDS via async global->LDS DMA (ASYNCcnt).
__global__ __launch_bounds__(128) void flash_attn(const __bf16* __restrict__ Q,
                                                  const __bf16* __restrict__ K,
                                                  const __bf16* __restrict__ Vt,
                                                  __bf16* __restrict__ O) {
  __shared__ __attribute__((aligned(32))) __bf16 ktile[4][2][32][KPAD];
  __shared__ __attribute__((aligned(32))) __bf16 plds[4][16][32];
  const int w    = threadIdx.x >> 5;            // wave in block (0..3)
  const int lane = threadIdx.x & 31;
  const int l16  = lane & 15;
  const int hi   = lane >> 4;
  const int gwave = blockIdx.x * 4 + w;
  const int qt = gwave & (S / 16 - 1);          // 0..127
  const int bh = gwave >> 7;                    // b*NH + h
  const int h  = bh & (NH - 1);
  const int b  = bh >> 4;
  const int q0 = qt * 16;
  const float scale = 0.08838834764831845f;     // 1/sqrt(128)
  const float NEGF  = -1.0e9f;

  const __bf16* Qp  = Q  + (size_t)(b * S) * HID + h * HD;
  const __bf16* Kp  = K  + (size_t)(b * S) * HID + h * HD;
  const __bf16* Vtp = Vt + (size_t)bh * HD * S;

  // LDS byte offsets of this wave's two K buffers (flat ptr low 32 bits == LDS addr)
  unsigned int kbase[2];
  kbase[0] = (unsigned int)(size_t)&ktile[w][0][0][0];
  kbase[1] = (unsigned int)(size_t)&ktile[w][1][0][0];

  // async prefetch of one 32x128 bf16 K tile: 512 x 16B chunks, 16 per lane
  auto prefetch_k = [&](int kv0, int bb) {
#pragma unroll
    for (int i = 0; i < 16; i++) {
      int chunk = lane + 32 * i;                // 0..511
      int row   = chunk >> 4;                   // 0..31
      int cc    = chunk & 15;                   // 16B chunk within row
      unsigned int lds = kbase[bb] + (unsigned int)(row * (KPAD * 2) + cc * 16);
      unsigned long long ga =
          (unsigned long long)(size_t)(Kp + (size_t)(kv0 + row) * HID) +
          (unsigned long long)(cc * 16);
      asm volatile("global_load_async_to_lds_b128 %0, %1, off"
                   :: "v"(lds), "v"(ga) : "memory");
    }
  };

  // Q A-fragments: 4 chunks of K(=d) 32
  v16bf qa[4];
  {
    const __bf16* qrow = Qp + (size_t)(q0 + l16) * HID;
#pragma unroll
    for (int c = 0; c < 4; c++) {
      v8bf a0 = *(const v8bf*)(qrow + c * 32 + 8 * hi);
      v8bf a1 = *(const v8bf*)(qrow + c * 32 + 16 + 8 * hi);
      qa[c] = cat8(a0, a1);
    }
  }

  v8f zero = {0.f,0.f,0.f,0.f,0.f,0.f,0.f,0.f};
  v8f acc[8] = {zero, zero, zero, zero, zero, zero, zero, zero};
  float mrow[8], lrow[8];
#pragma unroll
  for (int r = 0; r < 8; r++) { mrow[r] = -__builtin_inff(); lrow[r] = 0.f; }

  const int nkb = (q0 + 15) / 32 + 1;           // causal: kv blocks needed
  prefetch_k(0, 0);
  for (int kb = 0; kb < nkb; kb++) {
    const int kv0 = kb * 32;
    const int bb  = kb & 1;
    if (kb + 1 < nkb) {
      // overwrites buffer bb^1: its DS reads retired at last iteration's
      // s_wait_dscnt 0 (P reload), so WAR is safe.
      prefetch_k(kv0 + 32, bb ^ 1);
      asm volatile("s_wait_asynccnt 16" ::: "memory");  // current buffer ready
    } else {
      asm volatile("s_wait_asynccnt 0" ::: "memory");
    }
    // ---- scores from LDS K tile: two 16x16 tiles, 4 WMMA each
    const __bf16* kt = &ktile[w][bb][0][0];
    v8f sc[2] = {zero, zero};
#pragma unroll
    for (int t = 0; t < 2; t++) {
      const __bf16* krow = kt + (size_t)(16 * t + l16) * KPAD + 8 * hi;
#pragma unroll
      for (int c = 0; c < 4; c++) {
        v8bf b0 = *(const v8bf*)(krow + c * 32);
        v8bf b1 = *(const v8bf*)(krow + c * 32 + 16);
        sc[t] = wmma_bf16(qa[c], cat8(b0, b1), sc[t]);
      }
    }
    // ---- scale + causal mask + online softmax over 32 columns
    float p0v[8], p1v[8], bm[8];
#pragma unroll
    for (int r = 0; r < 8; r++) {
      int row = q0 + r + 8 * hi;
      float v0 = sc[0][r] * scale + ((kv0 + l16)      <= row ? 0.f : NEGF);
      float v1 = sc[1][r] * scale + ((kv0 + 16 + l16) <= row ? 0.f : NEGF);
      p0v[r] = v0; p1v[r] = v1;
      float m = fmaxf(v0, v1);
#pragma unroll
      for (int off = 1; off < 16; off <<= 1)
        m = fmaxf(m, __shfl_xor(m, off, 32));
      bm[r] = m;
    }
#pragma unroll
    for (int r = 0; r < 8; r++) {
      float mnew = fmaxf(mrow[r], bm[r]);
      float f  = __expf(mrow[r] - mnew);
      float p0 = __expf(p0v[r] - mnew);
      float p1 = __expf(p1v[r] - mnew);
      float rs = p0 + p1;
#pragma unroll
      for (int off = 1; off < 16; off <<= 1)
        rs += __shfl_xor(rs, off, 32);
      lrow[r] = lrow[r] * f + rs;
      mrow[r] = mnew;
#pragma unroll
      for (int t8 = 0; t8 < 8; t8++) acc[t8][r] *= f;
      // stage P in LDS (C-layout -> memory [m][n]) for A-fragment reload
      plds[w][r + 8 * hi][l16]      = (__bf16)p0;
      plds[w][r + 8 * hi][16 + l16] = (__bf16)p1;
    }
    asm volatile("s_wait_dscnt 0" ::: "memory");   // wave-local LDS visibility
    // ---- reload P as A fragment (16x32, k = kv-kv0)
    v8bf pp0 = *(const v8bf*)&plds[w][l16][8 * hi];
    v8bf pp1 = *(const v8bf*)&plds[w][l16][16 + 8 * hi];
    v16bf pf = cat8(pp0, pp1);
    // ---- P * V : 8 output d-tiles, B-frag contiguous from V^T
#pragma unroll
    for (int t8 = 0; t8 < 8; t8++) {
      v16bf vf = cat8(*(const v8bf*)(Vtp + (size_t)(16 * t8 + l16) * S + kv0 + 8 * hi),
                      *(const v8bf*)(Vtp + (size_t)(16 * t8 + l16) * S + kv0 + 16 + 8 * hi));
      acc[t8] = wmma_bf16(pf, vf, acc[t8]);
    }
  }
  // ---- normalize and store
#pragma unroll
  for (int r = 0; r < 8; r++) {
    float inv = 1.f / lrow[r];
#pragma unroll
    for (int t8 = 0; t8 < 8; t8++) {
      size_t row = (size_t)(b * S + q0 + r + 8 * hi);
      O[row * HID + h * HD + 16 * t8 + l16] = (__bf16)(acc[t8][r] * inv);
    }
  }
}

// ---------------------------------------------------------------- launch
extern "C" void kernel_launch(void* const* d_in, const int* in_sizes, int n_in,
                              void* d_out, int out_size, void* d_ws, size_t ws_size,
                              hipStream_t stream) {
  const float* hidden = (const float*)d_in[0];
  // d_in[1] = attention_mask (unused; causal mask computed inline)
  const float* cosp = (const float*)d_in[2];
  const float* sinp = (const float*)d_in[3];
  const float* wq = (const float*)d_in[4];
  const float* wk = (const float*)d_in[5];
  const float* wv = (const float*)d_in[6];
  const float* wo = (const float*)d_in[7];

  const size_t XE = (size_t)M_ROWS * HID;   // 8,388,608 activation elems
  const size_t WE = (size_t)HID * HID;      // 4,194,304 weight elems

  char* p = (char*)d_ws;
  __bf16* Xb  = (__bf16*)p; p += XE * 2;
  __bf16* Wqb = (__bf16*)p; p += WE * 2;
  __bf16* Wkb = (__bf16*)p; p += WE * 2;
  __bf16* Wvb = (__bf16*)p; p += WE * 2;
  __bf16* Wob = (__bf16*)p; p += WE * 2;
  __bf16* Qb  = (__bf16*)p; p += XE * 2;
  __bf16* Kb  = (__bf16*)p; p += XE * 2;
  __bf16* Vb  = (__bf16*)p; p += XE * 2;
  __bf16* Vt  = (__bf16*)p; p += XE * 2;
  __bf16* Ob  = Vb;                          // V dead after transpose; reuse

  // 1) fp32 -> bf16
  cvt_f32_bf16<<<2048, 256, 0, stream>>>(hidden, Xb, XE);
  cvt_f32_bf16<<<2048, 256, 0, stream>>>(wq, Wqb, WE);
  cvt_f32_bf16<<<2048, 256, 0, stream>>>(wk, Wkb, WE);
  cvt_f32_bf16<<<2048, 256, 0, stream>>>(wv, Wvb, WE);
  cvt_f32_bf16<<<2048, 256, 0, stream>>>(wo, Wob, WE);

  // 2) Q/K/V projections (WMMA): (4096/16)*(2048/128)=4096 waves -> 512 blocks
  gemm_xwt<__bf16><<<512, 256, 0, stream>>>(Xb, Wqb, Qb);
  gemm_xwt<__bf16><<<512, 256, 0, stream>>>(Xb, Wkb, Kb);
  gemm_xwt<__bf16><<<512, 256, 0, stream>>>(Xb, Wvb, Vb);

  // 3) RoPE on Q,K:  B*S*NH*HALF pair-threads
  rope_qk<<<(B * S * NH * HALF) / 256, 256, 0, stream>>>(Qb, Kb, cosp, sinp);

  // 4) V -> V^T
  transpose_v<<<dim3(S / 32, HD / 32, B * NH), 256, 0, stream>>>(Vb, Vt);

  // 5) flash attention: B*NH*(S/16) = 4096 waves, 4 waves/block -> 1024 blocks
  flash_attn<<<1024, 128, 0, stream>>>(Qb, Kb, Vt, Ob);

  // 6) output projection, fp32 out
  gemm_xwt<float><<<512, 256, 0, stream>>>(Ob, Wob, (float*)d_out);
}
